// GraphAttentionLayer_41540923687618
// MI455X (gfx1250) — matrix operand order repaired and checked by reading, not compiled, not run on hardware
//
#include <hip/hip_runtime.h>
#include <hip/hip_bf16.h>

typedef _Float16 f16;
typedef __attribute__((ext_vector_type(16))) _Float16 v16h;
typedef __attribute__((ext_vector_type(8)))  _Float16 v8h;
typedef __attribute__((ext_vector_type(4)))  _Float16 v4h;
typedef __attribute__((ext_vector_type(8)))  float    v8f;
typedef __attribute__((ext_vector_type(4)))  float    v4f;

#define NN   8192
#define INF  512
#define OUTF 128
#define LOG2E 1.44269504088896f

// ---------------------------------------------------------------------------
// Kernel 1: W [512][128] f32  ->  Wt16 [128][512] f16 (transposed, B-operand
// friendly: each output column's K values contiguous).
// ---------------------------------------------------------------------------
__global__ __launch_bounds__(256) void k_w16(const float* __restrict__ W,
                                             f16* __restrict__ Wt) {
  int idx = blockIdx.x * 256 + threadIdx.x;      // 65536 elements
  int k = idx >> 7, c = idx & 127;
  Wt[c * INF + k] = (f16)W[idx];
}

// ---------------------------------------------------------------------------
// Kernel 2: h = X @ W via WMMA f16->f32.  One block = 16 rows, 8 waves cover
// the 8 column tiles (128 cols).  Emits h16t [128][8192] (transposed) and
// h16 [8192][128] (row-major) in f16.
// ---------------------------------------------------------------------------
__global__ __launch_bounds__(256) void k_h(const float* __restrict__ X,
                                           const f16* __restrict__ Wt,
                                           f16* __restrict__ h16t,
                                           f16* __restrict__ h16) {
  __shared__ __align__(32) f16 xa[16 * 520];   // 520 halves/row: 16B aligned, 4-dword bank skew
  __shared__ __align__(32) f16 hT[128 * 24];   // C tiles transposed [col][row]
  const int t  = threadIdx.x;
  const int r0 = blockIdx.x * 16;

  // Stage X tile (16 x 512) as f16 into LDS, coalesced.
  #pragma unroll
  for (int i = 0; i < 32; ++i) {
    int idx = i * 256 + t;
    int row = idx >> 9, k = idx & 511;
    xa[row * 520 + k] = (f16)X[(size_t)(r0 + row) * INF + k];
  }
  __syncthreads();

  const int L = t & 31, wv = t >> 5;
  const int col  = wv * 16 + (L & 15);
  const int rowA = L & 15;
  const int ksel = L >> 4;

  v8f c = {};
  const f16* bp = Wt + (size_t)col * INF + ksel * 16;
  #pragma unroll
  for (int kb = 0; kb < 16; ++kb) {
    int k0 = kb * 32;
    const f16* ap = xa + rowA * 520 + k0 + ksel * 16;
    v8h alo = *(const v8h*)(ap);
    v8h ahi = *(const v8h*)(ap + 8);
    v16h a = __builtin_shufflevector(alo, ahi, 0,1,2,3,4,5,6,7,8,9,10,11,12,13,14,15);
    v16h b = *(const v16h*)(bp + k0);
    c = __builtin_amdgcn_wmma_f32_16x16x32_f16(false, a, false, b, (short)0, c, false, false);
  }

  // C layout: VGPR p, lanes 0-15 -> M=p, lanes 16-31 -> M=p+8, N = lane&15.
  #pragma unroll
  for (int p = 0; p < 8; ++p) {
    int m = p + ksel * 8;
    hT[col * 24 + m] = (f16)c[p];
  }
  __syncthreads();

  if (t < 128) {                                   // h16t[c][r0..r0+15]
    const f16* src = hT + t * 24;
    f16* dst = h16t + (size_t)t * NN + r0;
    *(v8h*)(dst)     = *(const v8h*)(src);
    *(v8h*)(dst + 8) = *(const v8h*)(src + 8);
  }
  {                                                // h16[r0+j][cs..cs+7]
    int j = t >> 4, cs = (t & 15) * 8;
    v8h v;
    #pragma unroll
    for (int q = 0; q < 8; ++q) v[q] = hT[(cs + q) * 24 + j];
    *(v8h*)(h16 + (size_t)(r0 + j) * OUTF + cs) = v;
  }
}

// ---------------------------------------------------------------------------
// Kernel 3: Wh1 = h @ a1, Wh2 = h @ a2.  One wave per row, shuffle reduce.
// ---------------------------------------------------------------------------
__global__ __launch_bounds__(256) void k_wh(const f16* __restrict__ h16,
                                            const float* __restrict__ a1,
                                            const float* __restrict__ a2,
                                            float* __restrict__ Wh1,
                                            float* __restrict__ Wh2) {
  int t = threadIdx.x, L = t & 31;
  int row = blockIdx.x * 8 + (t >> 5);
  v4h hv  = *(const v4h*)(h16 + (size_t)row * OUTF + L * 4);
  v4f a1v = *(const v4f*)(a1 + L * 4);
  v4f a2v = *(const v4f*)(a2 + L * 4);
  float s1 = 0.f, s2 = 0.f;
  #pragma unroll
  for (int q = 0; q < 4; ++q) {
    float hx = (float)hv[q];
    s1 += hx * a1v[q];
    s2 += hx * a2v[q];
  }
  #pragma unroll
  for (int off = 16; off > 0; off >>= 1) {
    s1 += __shfl_down(s1, off, 32);
    s2 += __shfl_down(s2, off, 32);
  }
  if (L == 0) { Wh1[row] = s1; Wh2[row] = s2; }
}

// ---------------------------------------------------------------------------
// Kernel 4: fused masked-softmax aggregation  out = softmax(mask) @ h.
// Block = 32 rows, 8 waves (each wave: one 16-col tile x two 16-row tiles).
// K chunks of 128: producer phase builds f16 weight tile in LDS, consumer
// phase runs WMMAs against L2-resident h16t.  adj prefetched per chunk.
// ---------------------------------------------------------------------------
__global__ __launch_bounds__(256) void k_attn(const int* __restrict__ adj,
                                              const f16* __restrict__ h16t,
                                              const float* __restrict__ Wh1,
                                              const float* __restrict__ Wh2,
                                              float* __restrict__ out) {
  __shared__ __align__(32) f16 wt[32 * 136];   // stride 136 halves = 68 dwords (bank skew 4)
  __shared__ float wh1s[32];
  __shared__ float rsum[32];
  const int t  = threadIdx.x;
  const int r0 = blockIdx.x * 32;
  if (t < 32) { wh1s[t] = Wh1[r0 + t]; rsum[t] = 0.f; }
  __syncthreads();

  const int rb = t >> 7;                 // 0/1: producer row interleave
  const int kk = t & 127;                // producer K lane within chunk
  const int L = t & 31, wv = t >> 5;
  const int colB = wv * 16 + (L & 15);
  const int hsel = L >> 4;
  const int rowA = L & 15;
  const f16* bp = h16t + (size_t)colB * NN + hsel * 16;

  float partial[16];
  #pragma unroll
  for (int i = 0; i < 16; ++i) partial[i] = 0.f;

  v8f c0 = {}, c1 = {};

  int adjv[16];
  float wh2v;
  #pragma unroll
  for (int rr = 0; rr < 16; ++rr)        // prefetch chunk 0 (nontemporal: streamed once)
    adjv[rr] = __builtin_nontemporal_load(adj + (size_t)(r0 + rr * 2 + rb) * NN + kk);
  wh2v = Wh2[kk];

  const int NCH = NN / 128;              // 64 chunks
  for (int ch = 0; ch < NCH; ++ch) {
    // ---- producer: w = adj>0 ? exp(sigmoid(Wh1_i + Wh2_j)) : 0 ----
    #pragma unroll
    for (int rr = 0; rr < 16; ++rr) {
      int rl = rr * 2 + rb;
      float x = wh1s[rl] + wh2v;
      float e = __builtin_amdgcn_exp2f(-LOG2E * x);
      float s = __builtin_amdgcn_rcpf(1.f + e);
      float w = __builtin_amdgcn_exp2f(LOG2E * s);
      w = adjv[rr] > 0 ? w : 0.f;
      partial[rr] += w;
      wt[rl * 136 + kk] = (f16)w;
    }
    __syncthreads();

    // ---- prefetch next adj chunk while WMMAs run ----
    if (ch + 1 < NCH) {
      int kg = (ch + 1) * 128 + kk;
      #pragma unroll
      for (int rr = 0; rr < 16; ++rr)
        adjv[rr] = __builtin_nontemporal_load(adj + (size_t)(r0 + rr * 2 + rb) * NN + kg);
      wh2v = Wh2[kg];
    }

    // ---- consumer: 4 K-steps of 32, two M tiles sharing each B fragment ----
    #pragma unroll
    for (int ks = 0; ks < 4; ++ks) {
      int kl = ks * 32;
      v16h b = *(const v16h*)(bp + ch * 128 + kl);
      const f16* a0p = wt + rowA * 136 + kl + hsel * 16;
      v8h a0lo = *(const v8h*)(a0p);
      v8h a0hi = *(const v8h*)(a0p + 8);
      v16h a0 = __builtin_shufflevector(a0lo, a0hi, 0,1,2,3,4,5,6,7,8,9,10,11,12,13,14,15);
      c0 = __builtin_amdgcn_wmma_f32_16x16x32_f16(false, a0, false, b, (short)0, c0, false, false);
      const f16* a1p = a0p + 16 * 136;
      v8h a1lo = *(const v8h*)(a1p);
      v8h a1hi = *(const v8h*)(a1p + 8);
      v16h a1 = __builtin_shufflevector(a1lo, a1hi, 0,1,2,3,4,5,6,7,8,9,10,11,12,13,14,15);
      c1 = __builtin_amdgcn_wmma_f32_16x16x32_f16(false, a1, false, b, (short)0, c1, false, false);
    }
    __syncthreads();
  }

  // ---- row-sum reduction: shuffle within wave, ds_add across waves ----
  #pragma unroll
  for (int rr = 0; rr < 16; ++rr) {
    float v = partial[rr];
    #pragma unroll
    for (int off = 16; off > 0; off >>= 1) v += __shfl_down(v, off, 32);
    if (L == 0) atomicAdd(&rsum[rr * 2 + rb], v);
  }
  __syncthreads();

  // ---- normalize and store ----
  #pragma unroll
  for (int p = 0; p < 8; ++p) {
    int ml = p + hsel * 8;
    float inv0 = __builtin_amdgcn_rcpf(fmaxf(rsum[ml],      1e-30f));
    float inv1 = __builtin_amdgcn_rcpf(fmaxf(rsum[ml + 16], 1e-30f));
    out[(size_t)(r0 + ml)      * OUTF + colB] = c0[p] * inv0;
    out[(size_t)(r0 + ml + 16) * OUTF + colB] = c1[p] * inv1;
  }
}

// ---------------------------------------------------------------------------
extern "C" void kernel_launch(void* const* d_in, const int* in_sizes, int n_in,
                              void* d_out, int out_size, void* d_ws, size_t ws_size,
                              hipStream_t stream) {
  const float* X   = (const float*)d_in[0];   // [8192,512]
  const int*   adj = (const int*)  d_in[1];   // [8192,8192]
  const float* W   = (const float*)d_in[2];   // [512,128]
  const float* a1  = (const float*)d_in[3];   // [128,1]
  const float* a2  = (const float*)d_in[4];   // [128,1]
  float* out = (float*)d_out;                 // [8192,128]

  char* ws = (char*)d_ws;
  f16*   Wt16 = (f16*)(ws);                                  // 128*512*2  = 128 KB
  f16*   h16t = (f16*)(ws + 131072);                         // 128*8192*2 = 2 MB
  f16*   h16  = (f16*)(ws + 131072 + 2097152);               // 8192*128*2 = 2 MB
  float* Wh1  = (float*)(ws + 131072 + 2 * 2097152);         // 32 KB
  float* Wh2  = (float*)(ws + 131072 + 2 * 2097152 + 32768); // 32 KB

  k_w16 <<<256,  256, 0, stream>>>(W, Wt16);
  k_h   <<<512,  256, 0, stream>>>(X, Wt16, h16t, h16);
  k_wh  <<<1024, 256, 0, stream>>>(h16, a1, a2, Wh1, Wh2);
  k_attn<<<256,  256, 0, stream>>>(adj, h16t, Wh1, Wh2, out);
}
